// ProxyNCAUnstable_81011673137813
// MI455X (gfx1250) — matrix operand order repaired and checked by reading, not compiled, not run on hardware
//
#include <hip/hip_runtime.h>
#include <hip/hip_bf16.h>
#include <math.h>

typedef __attribute__((ext_vector_type(2)))  float    v2f;
typedef __attribute__((ext_vector_type(8)))  float    v8f;
typedef __attribute__((ext_vector_type(2)))  _Float16 v2h;
typedef __attribute__((ext_vector_type(8)))  _Float16 v8h;
typedef __attribute__((ext_vector_type(16))) _Float16 v16h;

#define SZ_EMBED 64

// ---------------------------------------------------------------------------
// Kernel 1: L2-normalize rows of X / proxies, scale by 3.0, and emit both the
// fp32 copy (for the exact target-distance kernel) and the split-f16
// representation (hi = f16(x), lo = f16(x - hi)) used by the WMMA GEMM.
// One wave per row of 64 floats -> each lane owns a float2.
// ---------------------------------------------------------------------------
__global__ __launch_bounds__(256) void pnca_normalize(
    const float* __restrict__ X, const float* __restrict__ P,
    float* __restrict__ Xn, float* __restrict__ Pn,
    _Float16* __restrict__ Xh, _Float16* __restrict__ Xl,
    _Float16* __restrict__ Ph, _Float16* __restrict__ Pl, int N, int C) {
  const int lane = threadIdx.x & 31;
  const int wave = threadIdx.x >> 5;
  const int row = blockIdx.x * 8 + wave;
  if (row >= N + C) return;

  const float* src;
  float* dst;
  _Float16* dstH;
  _Float16* dstL;
  if (row < N) {
    src = X + (size_t)row * SZ_EMBED;
    dst = Xn + (size_t)row * SZ_EMBED;
    dstH = Xh + (size_t)row * SZ_EMBED;
    dstL = Xl + (size_t)row * SZ_EMBED;
  } else {
    const int r = row - N;
    src = P + (size_t)r * SZ_EMBED;
    dst = Pn + (size_t)r * SZ_EMBED;
    dstH = Ph + (size_t)r * SZ_EMBED;
    dstL = Pl + (size_t)r * SZ_EMBED;
  }

  v2f v = ((const v2f*)src)[lane];
  float ss = v.x * v.x + v.y * v.y;
#pragma unroll
  for (int off = 16; off >= 1; off >>= 1) ss += __shfl_xor(ss, off, 32);
  const float scale = 3.0f * rsqrtf(ss);
  v.x *= scale;
  v.y *= scale;
  ((v2f*)dst)[lane] = v;

  const _Float16 hx = (_Float16)v.x;
  const _Float16 hy = (_Float16)v.y;
  const _Float16 lx = (_Float16)(v.x - (float)hx);
  const _Float16 ly = (_Float16)(v.y - (float)hy);
  v2h h;
  h.x = hx;
  h.y = hy;
  ((v2h*)dstH)[lane] = h;
  v2h l;
  l.x = lx;
  l.y = ly;
  ((v2h*)dstL)[lane] = l;
}

// ---------------------------------------------------------------------------
// Kernel 2: fused GEMM + sum-exp + row-sum of D, D = 18 - 2*G.
// G = Xn * Pn^T computed in split-f16 via V_WMMA_F32_16X16X32_F16:
//   G ~= Ah*Bh + Ah*Bl + Al*Bh   (lo*lo term ~2^-33, dropped)
// Since ||Xn_i|| = ||Pn_c|| = 3 exactly, D in [0,36] and exp(D) <= e^36,
// so the logsumexp needs no running-max subtraction.
//
// Block = 8 waves, one 32-row M-tile per block (two 16-row WMMA sub-tiles
// sharing each B fragment). Each wave strides over 16x16 N-tiles.
// 16-bit fragment k order used here (same for A and B), kh = lane/16:
//   A (16x32): row = lane%16; 16 contiguous halves at K = kh*16
//   B (32x16): col = lane%16; 16 contiguous halves at K = kh*16
// C/D slot r on lane l is element (M = r + 8*kh, N = l%16).
// ---------------------------------------------------------------------------
__global__ __launch_bounds__(256) void pnca_main(
    const _Float16* __restrict__ Xh, const _Float16* __restrict__ Xl,
    const _Float16* __restrict__ Ph, const _Float16* __restrict__ Pl,
    float* __restrict__ rowLSE, float* __restrict__ rowSumD, int C) {
  const int lane = threadIdx.x & 31;
  const int wave = threadIdx.x >> 5;
  const int l15 = lane & 15;
  const int kh = lane >> 4;
  const int m0 = blockIdx.x * 32;

  // Hoist split A fragments: 2 sub-tiles x 2 K-chunks (K = 64 = 2 * 32).
  v16h ahi[2][2], alo[2][2];
#pragma unroll
  for (int j = 0; j < 2; ++j) {
    const size_t rb = (size_t)(m0 + j * 16 + l15) * SZ_EMBED;
#pragma unroll
    for (int q = 0; q < 2; ++q) {
      const size_t base = rb + q * 32 + kh * 16;
      const v8h h0 = *(const v8h*)(Xh + base);
      const v8h h1 = *(const v8h*)(Xh + base + 8);
      ahi[j][q] = __builtin_shufflevector(h0, h1, 0, 1, 2, 3, 4, 5, 6, 7, 8,
                                          9, 10, 11, 12, 13, 14, 15);
      const v8h l0 = *(const v8h*)(Xl + base);
      const v8h l1 = *(const v8h*)(Xl + base + 8);
      alo[j][q] = __builtin_shufflevector(l0, l1, 0, 1, 2, 3, 4, 5, 6, 7, 8,
                                          9, 10, 11, 12, 13, 14, 15);
    }
  }

  float sx[2][8], sd[2][8];
#pragma unroll
  for (int j = 0; j < 2; ++j)
#pragma unroll
    for (int r = 0; r < 8; ++r) {
      sx[j][r] = 0.0f;
      sd[j][r] = 0.0f;
    }

  const int ntiles = C >> 4;
  for (int nt = wave; nt < ntiles; nt += 8) {
    const size_t cb = (size_t)((nt << 4) + l15) * SZ_EMBED;

    // Prefetch next tile (per-lane addresses cover every 64B of each row).
    const int ntn = nt + 8;
    if (ntn < ntiles) {
      const size_t pfe = (size_t)((ntn << 4) + l15) * SZ_EMBED + kh * 32;
      __builtin_prefetch(Ph + pfe, 0, 0);
      __builtin_prefetch(Pl + pfe, 0, 0);
    }

    // Split B fragments: 2 K-chunks, hi + lo. 32B contiguous per lane.
    v16h bhi[2], blo[2];
#pragma unroll
    for (int q = 0; q < 2; ++q) {
      const size_t e = cb + q * 32 + kh * 16;
      bhi[q] = *(const v16h*)(Ph + e);
      blo[q] = *(const v16h*)(Pl + e);
    }

    v8f c0 = {0.f, 0.f, 0.f, 0.f, 0.f, 0.f, 0.f, 0.f};
    v8f c1 = {0.f, 0.f, 0.f, 0.f, 0.f, 0.f, 0.f, 0.f};
#pragma unroll
    for (int q = 0; q < 2; ++q) {
      c0 = __builtin_amdgcn_wmma_f32_16x16x32_f16(false, ahi[0][q], false,
                                                  bhi[q], (short)0, c0, false,
                                                  false);
      c1 = __builtin_amdgcn_wmma_f32_16x16x32_f16(false, ahi[1][q], false,
                                                  bhi[q], (short)0, c1, false,
                                                  false);
      c0 = __builtin_amdgcn_wmma_f32_16x16x32_f16(false, ahi[0][q], false,
                                                  blo[q], (short)0, c0, false,
                                                  false);
      c1 = __builtin_amdgcn_wmma_f32_16x16x32_f16(false, ahi[1][q], false,
                                                  blo[q], (short)0, c1, false,
                                                  false);
      c0 = __builtin_amdgcn_wmma_f32_16x16x32_f16(false, alo[0][q], false,
                                                  bhi[q], (short)0, c0, false,
                                                  false);
      c1 = __builtin_amdgcn_wmma_f32_16x16x32_f16(false, alo[1][q], false,
                                                  bhi[q], (short)0, c1, false,
                                                  false);
    }

#pragma unroll
    for (int r = 0; r < 8; ++r) {
      const float d0 = fmaf(-2.0f, c0[r], 18.0f);
      const float d1 = fmaf(-2.0f, c1[r], 18.0f);
      sx[0][r] += __expf(d0);
      sx[1][r] += __expf(d1);
      sd[0][r] += d0;
      sd[1][r] += d1;
    }
  }

  // Reduce across the 16 lanes of each half (different N columns, same rows).
#pragma unroll
  for (int j = 0; j < 2; ++j)
#pragma unroll
    for (int r = 0; r < 8; ++r) {
#pragma unroll
      for (int off = 8; off >= 1; off >>= 1) {
        sx[j][r] += __shfl_xor(sx[j][r], off, 32);
        sd[j][r] += __shfl_xor(sd[j][r], off, 32);
      }
    }

  __shared__ float lsx[8][32];
  __shared__ float lsd[8][32];
  if (l15 == 0) {
#pragma unroll
    for (int j = 0; j < 2; ++j)
#pragma unroll
      for (int r = 0; r < 8; ++r) {
        const int row = j * 16 + r + kh * 8;
        lsx[wave][row] = sx[j][r];
        lsd[wave][row] = sd[j][r];
      }
  }
  __syncthreads();

  if (threadIdx.x < 32) {
    const int t = threadIdx.x;
    float ss = 0.0f, sdd = 0.0f;
#pragma unroll
    for (int w = 0; w < 8; ++w) {
      ss += lsx[w][t];
      sdd += lsd[w][t];
    }
    rowLSE[m0 + t] = __logf(ss);
    rowSumD[m0 + t] = sdd;
  }
}

// ---------------------------------------------------------------------------
// Kernel 3: D[i, T[i]] = 18 - 2 * dot(Xn[i], Pn[T[i]]), exact fp32.
// One wave per row.
// ---------------------------------------------------------------------------
__global__ __launch_bounds__(256) void pnca_target(
    const float* __restrict__ Xn, const float* __restrict__ Pn,
    const int* __restrict__ T, float* __restrict__ rowDt, int N) {
  const int lane = threadIdx.x & 31;
  const int wave = threadIdx.x >> 5;
  const int row = blockIdx.x * 8 + wave;
  if (row >= N) return;
  const int t = T[row];
  const v2f x = ((const v2f*)(Xn + (size_t)row * SZ_EMBED))[lane];
  const v2f p = ((const v2f*)(Pn + (size_t)t * SZ_EMBED))[lane];
  float dot = x.x * p.x + x.y * p.y;
#pragma unroll
  for (int off = 16; off >= 1; off >>= 1) dot += __shfl_xor(dot, off, 32);
  if (lane == 0) rowDt[row] = 18.0f - 2.0f * dot;
}

// ---------------------------------------------------------------------------
// Kernel 4: loss = mean_i [ LSE_i - ( wt*Dt_i + woff*(sumD_i - Dt_i) ) ].
// ---------------------------------------------------------------------------
__global__ __launch_bounds__(256) void pnca_final(
    const float* __restrict__ rowLSE, const float* __restrict__ rowSumD,
    const float* __restrict__ rowDt, float* __restrict__ out, int N,
    float wt, float woff) {
  __shared__ float red[256];
  float acc = 0.0f;
  for (int i = threadIdx.x; i < N; i += 256) {
    const float dt = rowDt[i];
    acc += rowLSE[i] - (wt * dt + woff * (rowSumD[i] - dt));
  }
  red[threadIdx.x] = acc;
  __syncthreads();
#pragma unroll
  for (int s = 128; s >= 1; s >>= 1) {
    if (threadIdx.x < s) red[threadIdx.x] += red[threadIdx.x + s];
    __syncthreads();
  }
  if (threadIdx.x == 0) out[0] = red[0] / (float)N;
}

extern "C" void kernel_launch(void* const* d_in, const int* in_sizes, int n_in,
                              void* d_out, int out_size, void* d_ws, size_t ws_size,
                              hipStream_t stream) {
  (void)n_in; (void)out_size; (void)ws_size;
  const float* X = (const float*)d_in[0];
  const int* T = (const int*)d_in[1];
  const float* P = (const float*)d_in[2];
  float* out = (float*)d_out;

  const int N = in_sizes[0] / SZ_EMBED;  // 8192 samples
  const int C = in_sizes[2] / SZ_EMBED;  // 8192 classes

  float* ws = (float*)d_ws;
  float* Xn = ws;                             // N * 64 f32
  float* Pn = Xn + (size_t)N * SZ_EMBED;      // C * 64 f32
  float* rowLSE = Pn + (size_t)C * SZ_EMBED;  // N
  float* rowSumD = rowLSE + N;                // N
  float* rowDt = rowSumD + N;                 // N
  _Float16* Xh = (_Float16*)(rowDt + N);      // N * 64 f16 (hi)
  _Float16* Xl = Xh + (size_t)N * SZ_EMBED;   // N * 64 f16 (lo)
  _Float16* Ph = Xl + (size_t)N * SZ_EMBED;   // C * 64 f16 (hi)
  _Float16* Pl = Ph + (size_t)C * SZ_EMBED;   // C * 64 f16 (lo)

  const float smoothing = 0.1f;
  const float wt = 1.0f - smoothing;
  const float woff = smoothing / (float)(C - 1);

  // 1) normalize + split-f16 encode (one wave per row, 8 rows per block)
  pnca_normalize<<<(N + C + 7) / 8, 256, 0, stream>>>(X, P, Xn, Pn, Xh, Xl,
                                                      Ph, Pl, N, C);
  // 2) fused split-f16 WMMA GEMM + sum-exp + row-sum (32 rows per block)
  pnca_main<<<N / 32, 256, 0, stream>>>(Xh, Xl, Ph, Pl, rowLSE, rowSumD, C);
  // 3) exact target-class distances (fp32)
  pnca_target<<<(N + 7) / 8, 256, 0, stream>>>(Xn, Pn, T, rowDt, N);
  // 4) final scalar loss
  pnca_final<<<1, 256, 0, stream>>>(rowLSE, rowSumD, rowDt, out, N, wt, woff);
}
